// VarLenSelfAttention_70643622084990
// MI455X (gfx1250) — matrix operand order repaired and checked by reading, not compiled, not run on hardware
//
#include <hip/hip_runtime.h>
#include <hip/hip_bf16.h>

#define HIDDEN    2048
#define NUM_HEADS 16
#define NUM_KVH   4
#define HEAD_DIM  128
#define WINDOW    512
#define BATCH     4
#define SEQLEN    2048
#define TOKENS    (BATCH * SEQLEN)      // 8192
#define QKV_N     (HIDDEN + 2 * NUM_KVH * HEAD_DIM)   // 3072
#define KV_DIM    (NUM_KVH * HEAD_DIM)  // 512

typedef __attribute__((ext_vector_type(16))) __bf16 v16bf;
typedef __attribute__((ext_vector_type(8)))  float  v8f;
typedef __attribute__((ext_vector_type(4)))  unsigned int u32x4;

union Frag16 { v16bf v; u32x4 q[2]; };

// ---- WMMA fragment loaders (16-bit, K-step 32) ------------------------------
// A 16x32 (MxK): lane<16 holds row M=lane, K {0..7, 16..23}; lane>=16 row M=lane-16, K {8..15, 24..31}
__device__ __forceinline__ v16bf load_a_frag(const __bf16* p0, int ld, int lane) {
    const int r = lane & 15;
    const int k = (lane & 16) ? 8 : 0;
    const __bf16* p = p0 + (size_t)r * ld + k;
    Frag16 f;
    f.q[0] = *reinterpret_cast<const u32x4*>(p);        // K = k .. k+7
    f.q[1] = *reinterpret_cast<const u32x4*>(p + 16);   // K = k+16 .. k+23
    return f.v;
}
// B 32x16 (KxN) from row-major N x K storage: lane<16 holds col N=lane, K {0..7, 16..23}; lane>=16 col N=lane-16, K {8..15, 24..31}
__device__ __forceinline__ v16bf load_b_frag(const __bf16* p0, int ld, int lane) {
    const int r = lane & 15;
    const int k = (lane & 16) ? 8 : 0;
    const __bf16* p = p0 + (size_t)r * ld + k;
    Frag16 f;
    f.q[0] = *reinterpret_cast<const u32x4*>(p);
    f.q[1] = *reinterpret_cast<const u32x4*>(p + 16);
    return f.v;
}

__device__ __forceinline__ v8f wmma_bf16(v16bf a, v16bf b, v8f c) {
    return __builtin_amdgcn_wmma_f32_16x16x32_bf16(false, a, false, b, (short)0, c, false, false);
}

// async global -> LDS copy of 16 bytes (gfx1250 ASYNCcnt path)
__device__ __forceinline__ void async_copy_b128(const __bf16* gaddr, __bf16* lds_ptr) {
    unsigned loff = (unsigned)(uintptr_t)lds_ptr;   // low 32 bits of shared aperture = LDS byte offset
    asm volatile("global_load_async_to_lds_b128 %0, %1, off" :: "v"(loff), "v"(gaddr) : "memory");
}
__device__ __forceinline__ void wait_async() {
    asm volatile("s_wait_asynccnt 0x0" ::: "memory");
}

// ---- f32 -> bf16 convert ----------------------------------------------------
__global__ void cvt_kernel(const float* __restrict__ in, __bf16* __restrict__ out, int n) {
    int i = blockIdx.x * blockDim.x + threadIdx.x;
    if (i < n) out[i] = (__bf16)in[i];
}

// transpose-convert: in (R x C) f32 row-major -> out (C x R) bf16 row-major
__global__ void wtrans_kernel(const float* __restrict__ in, __bf16* __restrict__ out, int R, int C) {
    int tid = blockIdx.x * blockDim.x + threadIdx.x;
    if (tid >= R * C) return;
    int r = tid % R;
    int c = tid / R;
    out[tid] = (__bf16)in[(size_t)r * C + c];
}

// ---- GEMM: C[M,N] f32 = A[M,K] bf16 x Bt[N,K]^T bf16 ------------------------
// 256 threads = 8 waves; block tile 256x128; wave tile 64x64 (16 WMMAs / K-step).
// K-tiles (256x32 A, 128x32 B) are double-buffered in LDS, filled with
// global_load_async_to_lds_b128 so the k+1 prefetch overlaps the k compute.
__global__ __launch_bounds__(256) void gemm_bf16_kernel(
    const __bf16* __restrict__ A, const __bf16* __restrict__ Bt,
    float* __restrict__ C, int M, int N, int K) {
    __shared__ __bf16 Abuf[2][256 * 32];   // 16 KB per buffer
    __shared__ __bf16 Bbuf[2][128 * 32];   //  8 KB per buffer

    const int tid  = threadIdx.x;
    const int lane = tid & 31;
    const int wave = tid >> 5;              // 0..7
    const int m0w  = (wave >> 1) * 64;      // wave tile inside block tile
    const int n0w  = (wave & 1) * 64;

    const __bf16* Ag = A  + (size_t)(blockIdx.y * 256) * K;
    const __bf16* Bg = Bt + (size_t)(blockIdx.x * 128) * K;

    // issue async loads of the 32-wide K-slice k0 into buffer bi
    auto issue = [&](int bi, int k0) {
#pragma unroll
        for (int i = 0; i < 4; ++i) {                 // A: 256 rows x 4 chunks of 16B
            int cid = tid + i * 256;
            int row = cid >> 2, c = cid & 3;
            async_copy_b128(Ag + (size_t)row * K + k0 + c * 8, &Abuf[bi][row * 32 + c * 8]);
        }
#pragma unroll
        for (int i = 0; i < 2; ++i) {                 // B: 128 rows x 4 chunks of 16B
            int cid = tid + i * 256;
            int row = cid >> 2, c = cid & 3;
            async_copy_b128(Bg + (size_t)row * K + k0 + c * 8, &Bbuf[bi][row * 32 + c * 8]);
        }
    };

    v8f acc[4][4] = {};
    const int nsteps = K / 32;
    issue(0, 0);
    for (int step = 0; step < nsteps; ++step) {
        const int cur = step & 1;
        wait_async();                 // my prefetch for buf[cur] is complete
        __syncthreads();              // everyone's prefetch is complete
        if (step + 1 < nsteps) issue(cur ^ 1, (step + 1) * 32);

        v16bf af[4], bfg[4];
#pragma unroll
        for (int i = 0; i < 4; ++i) af[i]  = load_a_frag(&Abuf[cur][(m0w + 16 * i) * 32], 32, lane);
#pragma unroll
        for (int j = 0; j < 4; ++j) bfg[j] = load_b_frag(&Bbuf[cur][(n0w + 16 * j) * 32], 32, lane);
#pragma unroll
        for (int i = 0; i < 4; ++i)
#pragma unroll
            for (int j = 0; j < 4; ++j)
                acc[i][j] = wmma_bf16(af[i], bfg[j], acc[i][j]);
        __syncthreads();              // all reads of buf[cur] done before it is refilled
    }

    const int m0 = blockIdx.y * 256 + m0w;
    const int n0 = blockIdx.x * 128 + n0w;
    const int cn = lane & 15;
    const int rb = (lane & 16) ? 8 : 0;
#pragma unroll
    for (int i = 0; i < 4; ++i)
#pragma unroll
        for (int j = 0; j < 4; ++j)
#pragma unroll
            for (int v = 0; v < 8; ++v)
                C[(size_t)(m0 + 16 * i + rb + v) * N + (n0 + 16 * j + cn)] = acc[i][j][v];
}

// ---- RoPE on Q and K (reads fused QKV f32, writes bf16) ---------------------
__global__ void rope_kernel(const float* __restrict__ qkv, const int* __restrict__ pos_ids,
                            __bf16* __restrict__ qb, __bf16* __restrict__ kb) {
    int tid = blockIdx.x * blockDim.x + threadIdx.x;   // TOKENS * (16+4)*64
    if (tid >= TOKENS * 1280) return;
    int t = tid / 1280;
    int r = tid - t * 1280;
    float pos = (float)pos_ids[t];
    const float* src; __bf16* dst; int j;
    if (r < 1024) { int h = r >> 6; j = r & 63;
        src = qkv + (size_t)t * QKV_N + h * HEAD_DIM;
        dst = qb  + (size_t)t * HIDDEN + h * HEAD_DIM;
    } else { r -= 1024; int h = r >> 6; j = r & 63;
        src = qkv + (size_t)t * QKV_N + HIDDEN + h * HEAD_DIM;
        dst = kb  + (size_t)t * KV_DIM + h * HEAD_DIM;
    }
    float inv_freq = __expf(-(float)j * 0.14391156f);  // ln(10000)/64
    float ang = pos * inv_freq, s, c;
    __sincosf(ang, &s, &c);
    float xe = src[2 * j], xo = src[2 * j + 1];
    dst[2 * j]     = (__bf16)(xe * c - xo * s);
    dst[2 * j + 1] = (__bf16)(xe * s + xo * c);
}

// ---- V: (t, kvh, d) f32 slice of QKV -> (b, kvh, d, s) bf16 -----------------
__global__ void vtrans_kernel(const float* __restrict__ qkv, __bf16* __restrict__ vt) {
    int tid = blockIdx.x * blockDim.x + threadIdx.x;   // B*KVH*D*S = 4,194,304
    if (tid >= BATCH * NUM_KVH * HEAD_DIM * SEQLEN) return;
    int s   = tid & (SEQLEN - 1);
    int d   = (tid >> 11) & (HEAD_DIM - 1);
    int kvh = (tid >> 18) & (NUM_KVH - 1);
    int b   = tid >> 20;
    int t   = b * SEQLEN + s;
    vt[tid] = (__bf16)qkv[(size_t)t * QKV_N + HIDDEN + KV_DIM + kvh * HEAD_DIM + d];
}

// ---- Flash attention, sliding window 512, GQA -------------------------------
// one wave per (batch, head, 16-query tile); 128 threads = 4 waves per block
__global__ __launch_bounds__(128) void attn_kernel(
    const __bf16* __restrict__ qb, const __bf16* __restrict__ kb,
    const __bf16* __restrict__ vt, __bf16* __restrict__ attn) {
    __shared__ __bf16 pbuf[4][16][32];
    const int lane = threadIdx.x & 31;
    const int wave = threadIdx.x >> 5;
    const int w    = blockIdx.x * 4 + wave;
    const int qblk = w & 127;               // SEQLEN/16
    const int h    = (w >> 7) & 15;
    const int b    = w >> 11;
    const int q0   = qblk * 16;
    const int tb   = b * SEQLEN;
    const int kvh  = h >> 2;
    const int half = lane >> 4;
    const int cn   = lane & 15;
    const float scale = 0.08838834764831845f;   // 1/sqrt(128)

    v16bf qf[4];
    const __bf16* qbase = qb + (size_t)(tb + q0) * HIDDEN + h * HEAD_DIM;
#pragma unroll
    for (int c = 0; c < 4; ++c) qf[c] = load_a_frag(qbase + 32 * c, HIDDEN, lane);

    v8f o[8] = {};
    float mrow[8], lrow[8];
#pragma unroll
    for (int v = 0; v < 8; ++v) { mrow[v] = -1e30f; lrow[v] = 0.f; }

    int kstart = q0 - WINDOW; if (kstart < 0) kstart = 0; kstart &= ~31;
    const int kend = q0 + 15;

    for (int kp = kstart; kp <= kend; kp += 32) {
        // ---- scores: two 16x16 tiles covering keys kp..kp+31
        v8f s0 = {}, s1 = {};
        const __bf16* kbase = kb + (size_t)(tb + kp) * KV_DIM + kvh * HEAD_DIM;
#pragma unroll
        for (int c = 0; c < 4; ++c) {
            v16bf k0f = load_b_frag(kbase + 32 * c, KV_DIM, lane);
            v16bf k1f = load_b_frag(kbase + (size_t)16 * KV_DIM + 32 * c, KV_DIM, lane);
            s0 = wmma_bf16(qf[c], k0f, s0);
            s1 = wmma_bf16(qf[c], k1f, s1);
        }
        const int key0 = kp + cn, key1 = kp + 16 + cn;
        // ---- mask + online softmax (rows split across lane halves)
#pragma unroll
        for (int v = 0; v < 8; ++v) {
            const int qi = q0 + half * 8 + v;
            bool a0 = (key0 <= qi) && (qi - key0 <= WINDOW);
            bool a1 = (key1 <= qi) && (qi - key1 <= WINDOW);
            s0[v] = a0 ? s0[v] * scale : -1e38f;
            s1[v] = a1 ? s1[v] * scale : -1e38f;
            float rmax = fmaxf(s0[v], s1[v]);
#pragma unroll
            for (int off = 8; off >= 1; off >>= 1)
                rmax = fmaxf(rmax, __shfl_xor(rmax, off, 32));
            float mnew = fmaxf(mrow[v], rmax);
            float corr = __expf(mrow[v] - mnew);
            mrow[v] = mnew;
            float p0 = __expf(s0[v] - mnew);
            float p1 = __expf(s1[v] - mnew);
            float rsum = p0 + p1;
#pragma unroll
            for (int off = 8; off >= 1; off >>= 1)
                rsum += __shfl_xor(rsum, off, 32);
            lrow[v] = lrow[v] * corr + rsum;
#pragma unroll
            for (int j = 0; j < 8; ++j) o[j][v] *= corr;
            pbuf[wave][half * 8 + v][cn]      = (__bf16)p0;
            pbuf[wave][half * 8 + v][16 + cn] = (__bf16)p1;
        }
        // ---- P (16x32) through LDS into A-fragment layout, then P x V
        v16bf pf = load_a_frag(&pbuf[wave][0][0], 32, lane);
        const __bf16* vbase = vt + ((size_t)(b * NUM_KVH + kvh) * HEAD_DIM) * SEQLEN + kp;
#pragma unroll
        for (int j = 0; j < 8; ++j) {
            v16bf vf = load_b_frag(vbase + (size_t)(16 * j) * SEQLEN, SEQLEN, lane);
            o[j] = wmma_bf16(pf, vf, o[j]);
        }
    }
    // ---- normalize + store (t, h, d) bf16
#pragma unroll
    for (int v = 0; v < 8; ++v) {
        float inv = 1.0f / fmaxf(lrow[v], 1e-20f);
        size_t row = (size_t)(tb + q0 + half * 8 + v) * HIDDEN + h * HEAD_DIM;
#pragma unroll
        for (int j = 0; j < 8; ++j)
            attn[row + 16 * j + cn] = (__bf16)(o[j][v] * inv);
    }
}

extern "C" void kernel_launch(void* const* d_in, const int* in_sizes, int n_in,
                              void* d_out, int out_size, void* d_ws, size_t ws_size,
                              hipStream_t stream) {
    const float* hidden  = (const float*)d_in[0];
    const float* q_w     = (const float*)d_in[1];
    const float* k_w     = (const float*)d_in[2];
    const float* v_w     = (const float*)d_in[3];
    const float* o_w     = (const float*)d_in[4];
    const int*   pos_ids = (const int*)d_in[5];
    float* out = (float*)d_out;
    char* ws = (char*)d_ws;

    // workspace layout (bytes)
    __bf16* X     = (__bf16*)(ws + 0);          // 8192x2048 bf16   33,554,432
    __bf16* WqkvT = (__bf16*)(ws + 33554432);   // 3072x2048 bf16   12,582,912
    __bf16* OwT   = (__bf16*)(ws + 46137344);   // 2048x2048 bf16    8,388,608
    float*  QKV   = (float* )(ws + 54525952);   // 8192x3072 f32   100,663,296
    __bf16* Qbf   = (__bf16*)(ws + 155189248);  // 8192x2048 bf16   33,554,432
    __bf16* Kbf   = (__bf16*)(ws + 188743680);  // 8192x512  bf16    8,388,608
    __bf16* Vt    = (__bf16*)(ws + 197132288);  // 4x4x128x2048 bf16 8,388,608
    __bf16* Attn  = (__bf16*)(ws + 205520896);  // 8192x2048 bf16   33,554,432

    // 1) convert hidden_states to bf16
    cvt_kernel<<<(TOKENS * HIDDEN) / 256, 256, 0, stream>>>(hidden, X, TOKENS * HIDDEN);

    // 2) transpose-convert weights: W (HIDDEN x N) -> W^T (N x HIDDEN) bf16
    wtrans_kernel<<<(HIDDEN * HIDDEN) / 256, 256, 0, stream>>>(q_w, WqkvT, HIDDEN, HIDDEN);
    wtrans_kernel<<<(HIDDEN * KV_DIM) / 256, 256, 0, stream>>>(k_w, WqkvT + (size_t)HIDDEN * HIDDEN, HIDDEN, KV_DIM);
    wtrans_kernel<<<(HIDDEN * KV_DIM) / 256, 256, 0, stream>>>(v_w, WqkvT + (size_t)(HIDDEN + KV_DIM) * HIDDEN, HIDDEN, KV_DIM);
    wtrans_kernel<<<(HIDDEN * HIDDEN) / 256, 256, 0, stream>>>(o_w, OwT, HIDDEN, HIDDEN);

    // 3) fused QKV GEMM: QKV = X @ [q_w k_w v_w]
    {
        dim3 grid(QKV_N / 128, TOKENS / 256);
        gemm_bf16_kernel<<<grid, 256, 0, stream>>>(X, WqkvT, QKV, TOKENS, QKV_N, HIDDEN);
    }

    // 4) RoPE on Q,K (writes bf16); transpose V to (b,kvh,d,s) bf16
    rope_kernel<<<(TOKENS * 1280) / 256, 256, 0, stream>>>(QKV, pos_ids, Qbf, Kbf);
    vtrans_kernel<<<(BATCH * NUM_KVH * HEAD_DIM * SEQLEN) / 256, 256, 0, stream>>>(QKV, Vt);

    // 5) sliding-window flash attention (one wave per 16 queries per head)
    {
        int total_waves = BATCH * NUM_HEADS * (SEQLEN / 16);   // 8192
        attn_kernel<<<total_waves / 4, 128, 0, stream>>>(Qbf, Kbf, Vt, Attn);
    }

    // 6) output projection -> d_out (f32)
    {
        dim3 grid(HIDDEN / 128, TOKENS / 256);
        gemm_bf16_kernel<<<grid, 256, 0, stream>>>(Attn, OwT, out, TOKENS, HIDDEN, HIDDEN);
    }
}
